// RSSM_49289044689578
// MI455X (gfx1250) — matrix operand order repaired and checked by reading, not compiled, not run on hardware
//
#include <hip/hip_runtime.h>
#include <hip/hip_bf16.h>

typedef unsigned int u32;
typedef __attribute__((ext_vector_type(16))) __bf16 v16bf;
typedef __attribute__((ext_vector_type(8)))  float  v8f;
typedef __attribute__((ext_vector_type(4)))  u32    v4u;
typedef __attribute__((ext_vector_type(4)))  u32    u32x4;
typedef __attribute__((ext_vector_type(8)))  int    i32x8;
typedef __attribute__((ext_vector_type(4)))  int    i32x4;

#define B_    64
#define T_    128
#define E_    1024
#define S_    1024
#define DET_  512
#define HH_   1024
#define CC    32
#define NB_   8
#define XPAD  1056   // 1028 -> 33*32
#define PPAD  1568   // 1538 -> 49*32
#define NBLK  32
#define TPB   256
#define NTHREADS (NBLK*TPB)
#define SMEM_ELEMS (16 * PPAD)   // 25088 bf16 = 50176 B, max A-slab (16 rows x 1568)

// ---- Tensor Data Mover availability (device pass only; arity differs by toolchain) ----
#if defined(__AMDGCN__) && defined(__has_builtin)
#  if __has_builtin(__builtin_amdgcn_tensor_load_to_lds) && __has_builtin(__builtin_amdgcn_s_wait_tensorcnt)
#    define HAVE_TDM 1
#  else
#    define HAVE_TDM 0
#  endif
#else
#  define HAVE_TDM 0
#endif

// ---------------- device helpers ----------------
__device__ __forceinline__ u32 hashu(u32 x) {
  x ^= x >> 16; x *= 0x7feb352dU;
  x ^= x >> 15; x *= 0x846ca68bU;
  x ^= x >> 16; return x;
}
__device__ __forceinline__ float u01(u32 x) { return (hashu(x) >> 8) * (1.0f / 16777216.0f); }

struct GemmDst {
  const float* bias;   // length N (or null)
  int act;             // 0 none, 1 ELU, 2 SiLU
  __bf16* db; int ldb; // bf16 dest (next layer A) or null
  float*  df1; int ldf1;
  float*  df2; int ldf2;
};

// Stage nrows x K bf16 rows [row0, row0+nrows) of Ag (leading dim lda) into LDS slab sA.
// Preferred path: one TDM 2D-tile descriptor issued by wave 0 (TENSORcnt-tracked DMA).
__device__ __forceinline__ void stage_rows(__bf16* sA, const __bf16* __restrict__ Ag,
                                           int lda, int row0, int nrows, int K) {
#if HAVE_TDM
  if ((threadIdx.x >> 5) == 0) {
    const unsigned long long ga =
        (unsigned long long)(const void*)(Ag + (size_t)row0 * lda);     // tile start (bytes)
    const u32 laddr = (u32)(unsigned long long)(void*)sA;               // LDS byte offset = addr[31:0]
    // D# group0: count=1 valid user descriptor | lds_addr | global_addr[56:0] | type=2
    u32x4 g0 = { 1u,
                 laddr,
                 (u32)(ga & 0xffffffffu),
                 (u32)((ga >> 32) & 0x01ffffffu) | (2u << 30) };
    // D# group1: data_size=1 (2B) | tensor_dim0=K | tensor_dim1=nrows |
    //            tile_dim0=K | tile_dim1=nrows | tile_dim2=0 | tensor_dim0_stride=lda
    i32x8 g1 = { (int)(1u << 16),
                 (int)(((u32)K & 0xffffu) << 16),
                 (int)((((u32)K >> 16) & 0xffffu) | (((u32)nrows & 0xffffu) << 16)),
                 (int)((((u32)nrows >> 16) & 0xffffu) | (((u32)K & 0xffffu) << 16)),
                 (int)((u32)nrows & 0xffffu),
                 (int)(u32)lda,
                 0, 0 };
    i32x4 gz = { 0, 0, 0, 0 };
#if __clang_major__ >= 23
    i32x8 gz8 = { 0, 0, 0, 0, 0, 0, 0, 0 };
    __builtin_amdgcn_tensor_load_to_lds(g0, g1, gz, gz, gz8, 0);
#else
    __builtin_amdgcn_tensor_load_to_lds(g0, g1, gz, gz, 0);
#endif
    __builtin_amdgcn_s_wait_tensorcnt(0);
  }
  __syncthreads();
#else
  const int cpr = K >> 3;                    // 16B chunks per row
  for (int idx = threadIdx.x; idx < nrows * cpr; idx += TPB) {
    const int r = idx / cpr, c = idx - r * cpr;
    *(v4u*)(sA + (size_t)r * K + c * 8) =
        *(const v4u*)(Ag + (size_t)(row0 + r) * lda + c * 8);
  }
  __syncthreads();
#endif
}

// One 16x16 tile: A rows come from LDS slab (16 x K), B from global W[K x N].
__device__ __forceinline__ v8f tile16(const __bf16* sA, int K,
                                      const __bf16* __restrict__ Wm, int N, int nt) {
  const int lane = threadIdx.x & 31;
  v8f acc = {0.f,0.f,0.f,0.f,0.f,0.f,0.f,0.f};
  const __bf16* ar = sA + (size_t)(lane & 15) * K + ((lane >> 4) << 3);
  const __bf16* br = Wm + (size_t)((lane >> 4) << 3) * N + nt * 16 + (lane & 15);
  for (int k = 0, ks = K >> 5; k < ks; ++k) {
    union { v16bf v; v4u q[2]; } a, b;
    a.q[0] = *(const v4u*)(ar);
    a.q[1] = *(const v4u*)(ar + 16);
    for (int s = 0; s < 16; ++s) b.v[s] = br[(size_t)(s < 8 ? s : s + 8) * N];
    __builtin_prefetch(br + (size_t)32 * N, 0, 1);
    acc = __builtin_amdgcn_wmma_f32_16x16x32_bf16(false, a.v, false, b.v,
                                                  (short)0, acc, false, false);
    ar += 32; br += (size_t)32 * N;
  }
  return acc;
}

// Two 16x16 tiles sharing one A fragment load per k-step (register blocking on N).
__device__ __forceinline__ void tile16x2(const __bf16* sA, int K,
                                         const __bf16* __restrict__ Wm, int N,
                                         int nt0, int nt1, v8f& acc0, v8f& acc1) {
  const int lane = threadIdx.x & 31;
  const __bf16* ar = sA + (size_t)(lane & 15) * K + ((lane >> 4) << 3);
  const __bf16* b0 = Wm + (size_t)((lane >> 4) << 3) * N + nt0 * 16 + (lane & 15);
  const __bf16* b1 = Wm + (size_t)((lane >> 4) << 3) * N + nt1 * 16 + (lane & 15);
  for (int k = 0, ks = K >> 5; k < ks; ++k) {
    union { v16bf v; v4u q[2]; } a, x, y;
    a.q[0] = *(const v4u*)(ar);
    a.q[1] = *(const v4u*)(ar + 16);
    for (int s = 0; s < 16; ++s) {
      x.v[s] = b0[(size_t)(s < 8 ? s : s + 8) * N];
      y.v[s] = b1[(size_t)(s < 8 ? s : s + 8) * N];
    }
    __builtin_prefetch(b0 + (size_t)32 * N, 0, 1);
    acc0 = __builtin_amdgcn_wmma_f32_16x16x32_bf16(false, a.v, false, x.v,
                                                   (short)0, acc0, false, false);
    acc1 = __builtin_amdgcn_wmma_f32_16x16x32_bf16(false, a.v, false, y.v,
                                                   (short)0, acc1, false, false);
    ar += 32; b0 += (size_t)32 * N; b1 += (size_t)32 * N;
  }
}

__device__ __forceinline__ void epi_store(v8f acc, int mt, int nt, const GemmDst& D) {
  const int lane = threadIdx.x & 31;
  const int col  = nt * 16 + (lane & 15);
  const int row0 = mt * 16 + ((lane >> 4) << 3);
  const float bs = D.bias ? D.bias[col] : 0.0f;
#pragma unroll
  for (int r = 0; r < 8; ++r) {
    float v = acc[r] + bs;
    if (D.act == 1)      v = (v > 0.0f) ? v : (__expf(v) - 1.0f);
    else if (D.act == 2) v = v / (1.0f + __expf(-v));
    const int row = row0 + r;
    if (D.db)  D.db [(size_t)row * D.ldb  + col] = (__bf16)v;
    if (D.df1) D.df1[(size_t)row * D.ldf1 + col] = v;
    if (D.df2) D.df2[(size_t)row * D.ldf2 + col] = v;
  }
}

// ---------------- params ----------------
struct RssmP {
  const float *embeds, *actions, *goals;
  const float *pre_b1, *pre_b2, *gru_bih, *gru_bhh;
  const float *pri_b1, *pri_b2, *pri_b3, *pri_b4;
  const float *post_b1, *post_b2, *post_b3;
  const __bf16 *w_pre1, *w_pre2, *w_ih, *w_hh;
  const __bf16 *w_pri1, *w_pri2, *w_pri3, *w_pri4;
  const __bf16 *w_post1, *w_post2, *w_post3;
  __bf16 *x_bf, *h1_bf, *h2_bf, *det_bf, *p_bf;
  __bf16 *pa, *pb, *pc, *qa, *qb;
  float *gx_f, *gh_f, *det_f, *postl_f;
  float *out_det, *out_stoch, *out_post, *out_prior;
  u32 *bar;
};

// ---------------- persistent kernel ----------------
__global__ __launch_bounds__(TPB) void rssm_persist(RssmP p) {
  __shared__ __attribute__((aligned(16))) __bf16 sA[SMEM_ELEMS];
  const int tid = blockIdx.x * TPB + threadIdx.x;
  const int w   = threadIdx.x >> 5;     // wave id in block
  u32 gen = 0;

  auto gbar = [&]() {
    __syncthreads();
    ++gen;
    if (threadIdx.x == 0) {
      __hip_atomic_fetch_add(p.bar, 1u, __ATOMIC_RELEASE, __HIP_MEMORY_SCOPE_AGENT);
      while (__hip_atomic_load(p.bar, __ATOMIC_ACQUIRE, __HIP_MEMORY_SCOPE_AGENT) < gen * (u32)NBLK)
        __builtin_amdgcn_s_sleep(1);
    }
    __syncthreads();
  };

  // 16-block GEMM half-phase: out[64,1024] = A[64,K] @ W[K,1024] (+bias, act), 2 tiles/wave
  auto gemm16 = [&](int lb, const __bf16* Ag, int lda, int K,
                    const __bf16* Wm, const GemmDst& D) {
    const int mt = lb >> 2;
    const int ntBase = (lb & 3) * 16;
    stage_rows(sA, Ag, lda, mt * 16, 16, K);
    v8f a0 = {0.f,0.f,0.f,0.f,0.f,0.f,0.f,0.f};
    v8f a1 = {0.f,0.f,0.f,0.f,0.f,0.f,0.f,0.f};
    tile16x2(sA, K, Wm, HH_, ntBase + w, ntBase + 8 + w, a0, a1);
    epi_store(a0, mt, ntBase + w, D);
    epi_store(a1, mt, ntBase + 8 + w, D);
  };

  // ----- prologue: stoch0 one-hot + stage t=0 inputs -----
  for (int i = tid; i < B_ * CC; i += NTHREADS) {
    const int b = i >> 5, c = i & 31;
    int idx = (int)(u01(0x9E3779B9u ^ (u32)i) * 32.0f); if (idx > 31) idx = 31;
    __bf16* d = p.x_bf + (size_t)b * XPAD + c * 32;
    for (int k = 0; k < 32; ++k) d[k] = (__bf16)((k == idx) ? 1.0f : 0.0f);
  }
  for (int i = tid; i < B_ * 4; i += NTHREADS) {
    const int b = i >> 2, j = i & 3;
    const float v = (j < 2) ? p.actions[(size_t)b * T_ * 2 + j]
                            : p.goals  [(size_t)b * T_ * 2 + (j - 2)];
    p.x_bf[(size_t)b * XPAD + 1024 + j] = (__bf16)v;
    if (j >= 2) p.p_bf[(size_t)b * PPAD + 1536 + (j - 2)] = (__bf16)v;
  }
  for (int i = tid; i < B_ * E_; i += NTHREADS) {
    const int b = i >> 10, e = i & 1023;
    p.p_bf[(size_t)b * PPAD + 512 + e] = (__bf16)p.embeds[(size_t)b * T_ * E_ + e];
  }
  gbar();

  for (int t = 0; t < T_; ++t) {
    // F1: pre1 (ELU): x[64,1056] @ [1056,512] -> h1. 128 tiles, 4/block, waves 0-3.
    {
      const int mt = blockIdx.x >> 3;
      const int ntBase = (blockIdx.x & 7) * 4;
      stage_rows(sA, p.x_bf, XPAD, mt * 16, 16, XPAD);
      if (w < 4) {
        GemmDst D{p.pre_b1, 1, p.h1_bf, DET_, nullptr, 0, nullptr, 0};
        epi_store(tile16(sA, XPAD, p.w_pre1, DET_, ntBase + w), mt, ntBase + w, D);
      }
    }
    gbar();
    // F2: pre2 (ELU): h1 @ [512,512] -> h2
    {
      const int mt = blockIdx.x >> 3;
      const int ntBase = (blockIdx.x & 7) * 4;
      stage_rows(sA, p.h1_bf, DET_, mt * 16, 16, DET_);
      if (w < 4) {
        GemmDst D{p.pre_b2, 1, p.h2_bf, DET_, nullptr, 0, nullptr, 0};
        epi_store(tile16(sA, DET_, p.w_pre2, DET_, ntBase + w), mt, ntBase + w, D);
      }
    }
    gbar();
    // F3: GRU — block n (<8) does both 64x64@64x192 GEMMs AND gate math (WG-local sync)
    if (blockIdx.x < NB_) {
      const int n = blockIdx.x;
      __bf16* sX = sA;            // 64 x 64
      __bf16* sH = sA + 4096;     // 64 x 64
      stage_rows(sX, p.h2_bf  + n * 64, DET_, 0, 64, 64);
      stage_rows(sH, p.det_bf + n * 64, DET_, 0, 64, 64);
      const int lane = threadIdx.x & 31;
      for (int tj = w; tj < 96; tj += 8) {
        const int which = tj >= 48;
        const int r2 = tj - 48 * which;
        const int mt = r2 / 12, nt = r2 % 12;
        const __bf16* sT = (which ? sH : sX) + (size_t)mt * 16 * 64;
        const __bf16* Wn = (which ? p.w_hh : p.w_ih) + (size_t)n * (64 * 192);
        v8f acc = tile16(sT, 64, Wn, 192, nt);
        float* Of = which ? p.gh_f : p.gx_f;
        const int col  = nt * 16 + (lane & 15);
        const int row0 = mt * 16 + ((lane >> 4) << 3);
#pragma unroll
        for (int rr = 0; rr < 8; ++rr)
          Of[(size_t)(row0 + rr) * 1536 + n * 192 + col] = acc[rr];
      }
      __syncthreads();   // gx/gh of this n visible within the WGP
      for (int i = threadIdx.x; i < B_ * 64; i += TPB) {
        const int b = i >> 6, d = i & 63;
        const size_t g0 = (size_t)b * 1536 + n * 192;
        const int bb = n * 192;
        const float xr = p.gx_f[g0 + d]       + p.gru_bih[bb + d];
        const float xz = p.gx_f[g0 + 64 + d]  + p.gru_bih[bb + 64 + d];
        const float xn = p.gx_f[g0 + 128 + d] + p.gru_bih[bb + 128 + d];
        const float hr = p.gh_f[g0 + d]       + p.gru_bhh[bb + d];
        const float hz = p.gh_f[g0 + 64 + d]  + p.gru_bhh[bb + 64 + d];
        const float hn = p.gh_f[g0 + 128 + d] + p.gru_bhh[bb + 128 + d];
        const float rg = 1.0f / (1.0f + __expf(-(xr + hr)));
        const float zg = 1.0f / (1.0f + __expf(-(xz + hz)));
        const float ng = tanhf(xn + rg * hn);
        const int j = n * 64 + d;
        const size_t di = (size_t)b * DET_ + j;
        const float h = (1.0f - zg) * ng + zg * p.det_f[di];
        p.det_f[di]  = h;
        p.det_bf[di] = (__bf16)h;
        p.p_bf[(size_t)b * PPAD + j] = (__bf16)h;
        p.out_det[((size_t)b * T_ + t) * DET_ + j] = h;
      }
    }
    gbar();
    // F4: pri1 (SiLU) on blocks 0-15 || post1 (ELU) on blocks 16-31
    if (blockIdx.x < 16) {
      GemmDst D{p.pri_b1, 2, p.pa, HH_, nullptr, 0, nullptr, 0};
      gemm16(blockIdx.x, p.det_bf, DET_, DET_, p.w_pri1, D);
    } else {
      GemmDst D{p.post_b1, 1, p.qa, HH_, nullptr, 0, nullptr, 0};
      gemm16(blockIdx.x - 16, p.p_bf, PPAD, PPAD, p.w_post1, D);
    }
    gbar();
    // F5: pri2 (SiLU) || post2 (ELU)
    if (blockIdx.x < 16) {
      GemmDst D{p.pri_b2, 2, p.pb, HH_, nullptr, 0, nullptr, 0};
      gemm16(blockIdx.x, p.pa, HH_, HH_, p.w_pri2, D);
    } else {
      GemmDst D{p.post_b2, 1, p.qb, HH_, nullptr, 0, nullptr, 0};
      gemm16(blockIdx.x - 16, p.qa, HH_, HH_, p.w_post2, D);
    }
    gbar();
    // F6: pri3 (SiLU) || post3 (linear -> post logits)
    if (blockIdx.x < 16) {
      GemmDst D{p.pri_b3, 2, p.pc, HH_, nullptr, 0, nullptr, 0};
      gemm16(blockIdx.x, p.pb, HH_, HH_, p.w_pri3, D);
    } else {
      GemmDst D{p.post_b3, 0, nullptr, 0, p.postl_f, S_,
                p.out_post + (size_t)t * S_, T_ * S_};
      gemm16(blockIdx.x - 16, p.qb, HH_, HH_, p.w_post3, D);
    }
    gbar();
    // F7: pri4 (linear -> prior logits) on blocks 0-15 || sampling + next-t staging on 16-31
    if (blockIdx.x < 16) {
      GemmDst D{p.pri_b4, 0, nullptr, 0, p.out_prior + (size_t)t * S_, T_ * S_, nullptr, 0};
      gemm16(blockIdx.x, p.pc, HH_, HH_, p.w_pri4, D);
    } else {
      const int gt = (blockIdx.x - 16) * TPB + threadIdx.x;   // 0..4095
      if (gt < B_ * CC) {
        const int b = gt >> 5, c = gt & 31;
        const float* L = p.postl_f + (size_t)b * S_ + c * 32;
        float mx = L[0];
        for (int k = 1; k < 32; ++k) mx = fmaxf(mx, L[k]);
        float pr[32]; float s = 0.f;
        for (int k = 0; k < 32; ++k) { pr[k] = __expf(L[k] - mx); s += pr[k]; }
        const float inv = 1.0f / s;
        float s2 = 0.f;
        for (int k = 0; k < 32; ++k) { pr[k] = 0.99f * pr[k] * inv + 0.01f / 32.0f; s2 += pr[k]; }
        const float inv2 = 1.0f / (s2 + 1e-8f);
        const float u = u01((u32)(t * 2048 + gt) ^ 0xA511E9B3u);
        float cdf = 0.f; int idx = 31;
        for (int k = 0; k < 32; ++k) { cdf += pr[k] * inv2; if (u < cdf) { idx = k; break; } }
        float*  os = p.out_stoch + ((size_t)b * T_ + t) * S_ + c * 32;
        __bf16* xs = p.x_bf + (size_t)b * XPAD + c * 32;
        for (int k = 0; k < 32; ++k) {
          const float o = (k == idx) ? 1.0f : 0.0f;
          os[k] = o; xs[k] = (__bf16)o;
        }
      }
      if (t + 1 < T_) {
        for (int i = gt; i < B_ * 4; i += 16 * TPB) {
          const int b = i >> 2, j = i & 3;
          const float v = (j < 2) ? p.actions[((size_t)b * T_ + (t + 1)) * 2 + j]
                                  : p.goals  [((size_t)b * T_ + (t + 1)) * 2 + (j - 2)];
          p.x_bf[(size_t)b * XPAD + 1024 + j] = (__bf16)v;
          if (j >= 2) p.p_bf[(size_t)b * PPAD + 1536 + (j - 2)] = (__bf16)v;
        }
        for (int i = gt; i < B_ * E_; i += 16 * TPB) {
          const int b = i >> 10, e = i & 1023;
          p.p_bf[(size_t)b * PPAD + 512 + e] =
              (__bf16)p.embeds[((size_t)b * T_ + (t + 1)) * E_ + e];
        }
      }
    }
    gbar();
  }
}

// ---------------- weight prep kernels ----------------
__global__ void k_convert_pad(const float* __restrict__ src, __bf16* __restrict__ dst,
                              int K, int N, int KP) {
  const int i = blockIdx.x * blockDim.x + threadIdx.x;
  if (i >= KP * N) return;
  const int k = i / N;
  dst[i] = (__bf16)((k < K) ? src[i] : 0.0f);
}
// (8,192,64) [n,g,d] -> (8,64,192) [n,d,g]  (B operand wants K(=d) x N(=g))
__global__ void k_gru_T(const float* __restrict__ src, __bf16* __restrict__ dst) {
  const int i = blockIdx.x * blockDim.x + threadIdx.x;
  if (i >= NB_ * 192 * 64) return;
  const int n = i / (192 * 64), rem = i % (192 * 64), g = rem / 64, d = rem % 64;
  dst[(size_t)n * (64 * 192) + d * 192 + g] = (__bf16)src[i];
}

// ---------------- host ----------------
extern "C" void kernel_launch(void* const* d_in, const int* in_sizes, int n_in,
                              void* d_out, int out_size, void* d_ws, size_t ws_size,
                              hipStream_t stream) {
  (void)in_sizes; (void)n_in; (void)out_size;
  const float* embeds  = (const float*)d_in[0];
  const float* actions = (const float*)d_in[1];
  const float* goals   = (const float*)d_in[2];
  const float* pre_w1  = (const float*)d_in[3];
  const float* pre_b1  = (const float*)d_in[4];
  const float* pre_w2  = (const float*)d_in[5];
  const float* pre_b2  = (const float*)d_in[6];
  const float* gru_wih = (const float*)d_in[7];
  const float* gru_whh = (const float*)d_in[8];
  const float* gru_bih = (const float*)d_in[9];
  const float* gru_bhh = (const float*)d_in[10];
  const float* pri_w1  = (const float*)d_in[11];
  const float* pri_b1  = (const float*)d_in[12];
  const float* pri_w2  = (const float*)d_in[13];
  const float* pri_b2  = (const float*)d_in[14];
  const float* pri_w3  = (const float*)d_in[15];
  const float* pri_b3  = (const float*)d_in[16];
  const float* pri_w4  = (const float*)d_in[17];
  const float* pri_b4  = (const float*)d_in[18];
  const float* post_w1 = (const float*)d_in[19];
  const float* post_b1 = (const float*)d_in[20];
  const float* post_w2 = (const float*)d_in[21];
  const float* post_b2 = (const float*)d_in[22];
  const float* post_w3 = (const float*)d_in[23];
  const float* post_b3 = (const float*)d_in[24];

  char* base = (char*)d_ws;
  size_t off = 0;
  auto alloc = [&](size_t bytes) -> void* {
    off = (off + 255) & ~(size_t)255;
    void* pv = base + off;
    off += bytes;
    return pv;
  };
  u32*    bar     = (u32*)   alloc(4);
  __bf16* w_pre1  = (__bf16*)alloc((size_t)XPAD * 512 * 2);
  __bf16* w_pre2  = (__bf16*)alloc((size_t)512 * 512 * 2);
  __bf16* w_ih    = (__bf16*)alloc((size_t)NB_ * 64 * 192 * 2);
  __bf16* w_hh    = (__bf16*)alloc((size_t)NB_ * 64 * 192 * 2);
  __bf16* w_pri1  = (__bf16*)alloc((size_t)512 * 1024 * 2);
  __bf16* w_pri2  = (__bf16*)alloc((size_t)1024 * 1024 * 2);
  __bf16* w_pri3  = (__bf16*)alloc((size_t)1024 * 1024 * 2);
  __bf16* w_pri4  = (__bf16*)alloc((size_t)1024 * 1024 * 2);
  __bf16* w_post1 = (__bf16*)alloc((size_t)PPAD * 1024 * 2);
  __bf16* w_post2 = (__bf16*)alloc((size_t)1024 * 1024 * 2);
  __bf16* w_post3 = (__bf16*)alloc((size_t)1024 * 1024 * 2);
  __bf16* x_bf    = (__bf16*)alloc((size_t)B_ * XPAD * 2);
  __bf16* h1_bf   = (__bf16*)alloc((size_t)B_ * DET_ * 2);
  __bf16* h2_bf   = (__bf16*)alloc((size_t)B_ * DET_ * 2);
  __bf16* det_bf  = (__bf16*)alloc((size_t)B_ * DET_ * 2);
  __bf16* p_bf    = (__bf16*)alloc((size_t)B_ * PPAD * 2);
  __bf16* pa      = (__bf16*)alloc((size_t)B_ * HH_ * 2);
  __bf16* pb      = (__bf16*)alloc((size_t)B_ * HH_ * 2);
  __bf16* pc      = (__bf16*)alloc((size_t)B_ * HH_ * 2);
  __bf16* qa      = (__bf16*)alloc((size_t)B_ * HH_ * 2);
  __bf16* qb      = (__bf16*)alloc((size_t)B_ * HH_ * 2);
  float*  gx_f    = (float*) alloc((size_t)B_ * 1536 * 4);
  float*  gh_f    = (float*) alloc((size_t)B_ * 1536 * 4);
  float*  det_f   = (float*) alloc((size_t)B_ * DET_ * 4);
  float*  postl_f = (float*) alloc((size_t)B_ * S_ * 4);
  if (off > ws_size) return;

  hipMemsetAsync(d_ws, 0, off, stream);   // barrier counter, padded tails, deter0

  auto conv = [&](const float* s, __bf16* d, int K, int N, int KP) {
    const int total = KP * N;
    k_convert_pad<<<(total + 255) / 256, 256, 0, stream>>>(s, d, K, N, KP);
  };
  conv(pre_w1, w_pre1, 1028, 512, XPAD);
  conv(pre_w2, w_pre2, 512, 512, 512);
  conv(pri_w1, w_pri1, 512, 1024, 512);
  conv(pri_w2, w_pri2, 1024, 1024, 1024);
  conv(pri_w3, w_pri3, 1024, 1024, 1024);
  conv(pri_w4, w_pri4, 1024, 1024, 1024);
  conv(post_w1, w_post1, 1538, 1024, PPAD);
  conv(post_w2, w_post2, 1024, 1024, 1024);
  conv(post_w3, w_post3, 1024, 1024, 1024);
  {
    const int total = NB_ * 192 * 64;
    k_gru_T<<<(total + 255) / 256, 256, 0, stream>>>(gru_wih, w_ih);
    k_gru_T<<<(total + 255) / 256, 256, 0, stream>>>(gru_whh, w_hh);
  }

  float* out = (float*)d_out;
  RssmP prm;
  prm.embeds = embeds; prm.actions = actions; prm.goals = goals;
  prm.pre_b1 = pre_b1; prm.pre_b2 = pre_b2;
  prm.gru_bih = gru_bih; prm.gru_bhh = gru_bhh;
  prm.pri_b1 = pri_b1; prm.pri_b2 = pri_b2; prm.pri_b3 = pri_b3; prm.pri_b4 = pri_b4;
  prm.post_b1 = post_b1; prm.post_b2 = post_b2; prm.post_b3 = post_b3;
  prm.w_pre1 = w_pre1; prm.w_pre2 = w_pre2; prm.w_ih = w_ih; prm.w_hh = w_hh;
  prm.w_pri1 = w_pri1; prm.w_pri2 = w_pri2; prm.w_pri3 = w_pri3; prm.w_pri4 = w_pri4;
  prm.w_post1 = w_post1; prm.w_post2 = w_post2; prm.w_post3 = w_post3;
  prm.x_bf = x_bf; prm.h1_bf = h1_bf; prm.h2_bf = h2_bf; prm.det_bf = det_bf;
  prm.p_bf = p_bf; prm.pa = pa; prm.pb = pb; prm.pc = pc; prm.qa = qa; prm.qb = qb;
  prm.gx_f = gx_f; prm.gh_f = gh_f; prm.det_f = det_f; prm.postl_f = postl_f;
  prm.out_det   = out;
  prm.out_stoch = out + (size_t)B_ * T_ * DET_;
  prm.out_post  = out + (size_t)B_ * T_ * (DET_ + S_);
  prm.out_prior = out + (size_t)B_ * T_ * (DET_ + 2 * S_);
  prm.bar = bar;

  rssm_persist<<<NBLK, TPB, 0, stream>>>(prm);
}
